// Encoder_12790412607643
// MI455X (gfx1250) — matrix operand order repaired and checked by reading, not compiled, not run on hardware
//
#include <hip/hip_runtime.h>
#include <math.h>

typedef _Float16 half_t;
typedef __attribute__((ext_vector_type(16))) _Float16 v16h;
typedef __attribute__((ext_vector_type(8)))  _Float16 v8h;
typedef __attribute__((ext_vector_type(8)))  float    v8f;

#define MAX_NORM 0.996f     /* 1.0 - 4e-3 */
#define EPSF     1e-15f
#define ATANH_CLIP (1.0f - 1e-7f)

#define WMMA_F16(a, b, c) \
    __builtin_amdgcn_wmma_f32_16x16x32_f16(false, (a), false, (b), (short)0, (c), false, false)

// ---------------------------------------------------------------------------
// block-wide sum reduction (blockDim.x == 256, returns broadcast value)
// ---------------------------------------------------------------------------
__device__ __forceinline__ float block_reduce_sum(float v, float* smem) {
    #pragma unroll
    for (int off = 16; off > 0; off >>= 1) v += __shfl_xor(v, off, 32);
    int wid  = threadIdx.x >> 5;
    int lane = threadIdx.x & 31;
    if (lane == 0) smem[wid] = v;
    __syncthreads();
    float r = (threadIdx.x < (blockDim.x >> 5)) ? smem[threadIdx.x] : 0.0f;
    if (wid == 0) {
        #pragma unroll
        for (int off = 4; off > 0; off >>= 1) r += __shfl_xor(r, off, 32);
        if (lane == 0) smem[0] = r;
    }
    __syncthreads();
    float out = smem[0];
    __syncthreads();            // safe for back-to-back reuse of smem
    return out;
}

// ---------------------------------------------------------------------------
// degree / normalization
// ---------------------------------------------------------------------------
__global__ void deg_init_kernel(float* deg, int n) {
    int i = blockIdx.x * blockDim.x + threadIdx.x;
    if (i < n) deg[i] = 1.0f;                      // self-loop contribution
}

__global__ void deg_accum_kernel(const int* __restrict__ col, float* deg, int E) {
    int i = blockIdx.x * blockDim.x + threadIdx.x;
    if (i < E) unsafeAtomicAdd(&deg[col[i]], 1.0f);
}

__global__ void deg_to_dis_kernel(float* deg, int n) {
    int i = blockIdx.x * blockDim.x + threadIdx.x;
    if (i < n) {
        float d = deg[i];
        deg[i] = (d > 0.0f) ? 1.0f / sqrtf(d) : 0.0f;
    }
}

// ---------------------------------------------------------------------------
// row-wise expmap0 (+ projection). src/dst may alias. dst16 optional.
// ---------------------------------------------------------------------------
__global__ __launch_bounds__(256)
void expmap_rows_kernel(const float* __restrict__ src, float* __restrict__ dst,
                        half_t* __restrict__ dst16, int D) {
    __shared__ float smem[16];
    size_t row = blockIdx.x;
    const float* s = src + row * (size_t)D;
    float part = 0.0f;
    for (int d = threadIdx.x; d < D; d += blockDim.x) { float v = s[d]; part += v * v; }
    float n2  = block_reduce_sum(part, smem);
    float nrm = sqrtf(n2);
    float nc  = fmaxf(nrm, EPSF);
    float sc  = tanhf(nc) / nc;
    float rn  = nrm * sc;                                  // resulting norm
    float pf  = (rn > MAX_NORM) ? MAX_NORM / fmaxf(rn, EPSF) : 1.0f;
    sc *= pf;
    float*  drow = dst + row * (size_t)D;
    half_t* hrow = dst16 ? (dst16 + row * (size_t)D) : (half_t*)0;
    for (int d = threadIdx.x; d < D; d += blockDim.x) {
        float v = s[d] * sc;
        drow[d] = v;
        if (hrow) hrow[d] = (half_t)v;
    }
}

// ---------------------------------------------------------------------------
// eb = expmap0(b); eb[D] = ||eb||^2  (single block)
// ---------------------------------------------------------------------------
__global__ __launch_bounds__(256)
void bias_expmap_kernel(const float* __restrict__ b, float* __restrict__ eb, int D) {
    __shared__ float smem[16];
    float part = 0.0f;
    for (int d = threadIdx.x; d < D; d += blockDim.x) { float v = b[d]; part += v * v; }
    float n2  = block_reduce_sum(part, smem);
    float nrm = sqrtf(n2);
    float nc  = fmaxf(nrm, EPSF);
    float sc  = tanhf(nc) / nc;
    float rn  = nrm * sc;
    float pf  = (rn > MAX_NORM) ? MAX_NORM / fmaxf(rn, EPSF) : 1.0f;
    sc *= pf;
    for (int d = threadIdx.x; d < D; d += blockDim.x) eb[d] = b[d] * sc;
    if (threadIdx.x == 0) eb[D] = n2 * sc * sc;            // ||eb||^2
}

// ---------------------------------------------------------------------------
// f32 -> f16 convert
// ---------------------------------------------------------------------------
__global__ void convert_f16_kernel(const float* __restrict__ src, half_t* __restrict__ dst, int count) {
    int i = blockIdx.x * blockDim.x + threadIdx.x;
    if (i < count) dst[i] = (half_t)src[i];
}

// ---------------------------------------------------------------------------
// WMMA GEMM: C[m,n] = sum_k A16[m,k] * B16[n,k]   (A: h16 rows, B: W16 rows)
// One wave computes a 32x32 output tile (2x2 register blocking -> 4 WMMAs per
// k-step with 2 A-frags + 2 B-frags: 2 b128 loads per WMMA).
// Fragment k order: contiguous k = 16*grp + s in A and B (16-bit A 16x32, B 32x16, f32 C/D).
// ---------------------------------------------------------------------------
__global__ __launch_bounds__(256)
void gemm_wmma_kernel(const half_t* __restrict__ A, const half_t* __restrict__ B,
                      float* __restrict__ C, int M, int K, int Nout,
                      int Ntiles2, int totalTiles2) {
    int wave = blockIdx.x * (blockDim.x >> 5) + (threadIdx.x >> 5);
    if (wave >= totalTiles2) return;                // wave-uniform: EXEC stays all-ones
    int lane = threadIdx.x & 31;
    int mt   = wave / Ntiles2;
    int nt   = wave % Ntiles2;
    int grp  = lane >> 4;                           // lane group 0 / 1
    int l16  = lane & 15;

    // fragment source rows (clamped for safety; invalid outputs never stored)
    int ar0 = mt * 32 + l16;       ar0 = (ar0 < M)    ? ar0 : M - 1;
    int ar1 = mt * 32 + 16 + l16;  ar1 = (ar1 < M)    ? ar1 : M - 1;
    int br0 = nt * 32 + l16;       br0 = (br0 < Nout) ? br0 : Nout - 1;
    int br1 = nt * 32 + 16 + l16;  br1 = (br1 < Nout) ? br1 : Nout - 1;

    const half_t* a0p = A + (size_t)ar0 * K;
    const half_t* a1p = A + (size_t)ar1 * K;
    const half_t* b0p = B + (size_t)br0 * K;
    const half_t* b1p = B + (size_t)br1 * K;

    v8f acc00 = {}, acc01 = {}, acc10 = {}, acc11 = {};
    for (int k0 = 0; k0 < K; k0 += 32) {
        // A 16x32 f16: grp0: K0..15 ; grp1: K16..31 (contiguous 16 halves)
        v16h a0 = *(const v16h*)(a0p + k0 + grp * 16);
        v16h a1 = *(const v16h*)(a1p + k0 + grp * 16);
        // B 32x16 f16: grp0: K0..15 ; grp1: K16..31 (contiguous 16 halves)
        v16h b0 = *(const v16h*)(b0p + k0 + grp * 16);
        v16h b1 = *(const v16h*)(b1p + k0 + grp * 16);

        acc00 = WMMA_F16(a0, b0, acc00);
        acc01 = WMMA_F16(a0, b1, acc01);
        acc10 = WMMA_F16(a1, b0, acc10);
        acc11 = WMMA_F16(a1, b1, acc11);
    }

    // C/D f32 16x16: lane l16 -> N, grp*8 + v -> M
    int n0 = nt * 32 + l16;
    int n1 = n0 + 16;
    int m0 = mt * 32 + grp * 8;
    int m1 = m0 + 16;
    bool okn0 = (n0 < Nout), okn1 = (n1 < Nout);
    #pragma unroll
    for (int v = 0; v < 8; ++v) {
        if (m0 + v < M) {
            if (okn0) C[(size_t)(m0 + v) * Nout + n0] = acc00[v];
            if (okn1) C[(size_t)(m0 + v) * Nout + n1] = acc01[v];
        }
        if (m1 + v < M) {
            if (okn0) C[(size_t)(m1 + v) * Nout + n0] = acc10[v];
            if (okn1) C[(size_t)(m1 + v) * Nout + n1] = acc11[v];
        }
    }
}

// ---------------------------------------------------------------------------
// Fused mobius_matvec-scaling + mobius_add(eb) + project + logmap0.
// Reads h32 (for xn) and mx; writes t in place over mx:  t = s1*mx + s2*eb
// ---------------------------------------------------------------------------
__global__ __launch_bounds__(256)
void mobius_post_kernel(const float* __restrict__ h32, float* __restrict__ mx,
                        const float* __restrict__ eb, int D) {
    __shared__ float smem[16];
    size_t row = blockIdx.x;
    const float* hrow = h32 + row * (size_t)D;
    float*       mrow = mx  + row * (size_t)D;

    float p_hh = 0.0f, p_mm = 0.0f, p_me = 0.0f;
    for (int d = threadIdx.x; d < D; d += blockDim.x) {
        float hv = hrow[d], mv = mrow[d], ev = eb[d];
        p_hh += hv * hv; p_mm += mv * mv; p_me += mv * ev;
    }
    float S_hh = block_reduce_sum(p_hh, smem);
    float S_mm = block_reduce_sum(p_mm, smem);
    float S_me = block_reduce_sum(p_me, smem);
    float Y2   = eb[D];

    // mobius_matvec tail: res = tanh(mxn/xn * artanh(xn)) * mx / mxn, projected
    float xn   = fmaxf(sqrtf(S_hh), EPSF);
    float mxn  = fmaxf(sqrtf(S_mm), EPSF);
    float xcl  = fminf(fmaxf(xn, -ATANH_CLIP), ATANH_CLIP);
    float tnh  = tanhf(mxn / xn * atanhf(xcl));
    float c    = tnh / mxn;
    float resn = fabsf(tnh) * (sqrtf(S_mm) / mxn);
    float pf1  = (resn > MAX_NORM) ? MAX_NORM / fmaxf(resn, EPSF) : 1.0f;
    c *= pf1;                                        // res = c * mx

    // mobius_add(res, eb): h' = a*res + g*eb
    float x2  = c * c * S_mm;
    float xy  = c * S_me;
    float den = fmaxf(1.0f + 2.0f * xy + x2 * Y2, EPSF);
    float a   = (1.0f + 2.0f * xy + Y2) / den;
    float g   = (1.0f - x2) / den;
    float ac  = a * c;                               // h' = ac*mx + g*eb

    // project(h') + logmap0
    float hn2 = ac * ac * S_mm + 2.0f * ac * g * S_me + g * g * Y2;
    float hn  = sqrtf(fmaxf(hn2, 0.0f));
    float pf2 = (hn > MAX_NORM) ? MAX_NORM / fmaxf(hn, EPSF) : 1.0f;
    float hpn = fmaxf(hn * pf2, EPSF);
    float lf  = atanhf(fminf(hpn, ATANH_CLIP)) / hpn;
    float s1  = lf * pf2 * ac;
    float s2  = lf * pf2 * g;

    for (int d = threadIdx.x; d < D; d += blockDim.x)
        mrow[d] = s1 * mrow[d] + s2 * eb[d];
}

// ---------------------------------------------------------------------------
// Edge scatter: agg[col] += dis[row]*dis[col] * t[row]   (one block per edge;
// edges >= E are self loops). float4 gathers + 4 fp atomics per thread.
// agg must be pre-zeroed. Requires D % 4 == 0.
// ---------------------------------------------------------------------------
__global__ __launch_bounds__(128)
void aggregate_kernel(const float* __restrict__ t, const int* __restrict__ row,
                      const int* __restrict__ col, const float* __restrict__ dis,
                      float* __restrict__ agg, int E, int D) {
    int e = blockIdx.x;
    int r, c;
    if (e < E) { r = row[e]; c = col[e]; } else { r = e - E; c = r; }
    float w = dis[r] * dis[c];
    const float4* trow = (const float4*)(t + (size_t)r * D);
    float*        arow = agg + (size_t)c * D;
    int nvec = D >> 2;
    for (int d = threadIdx.x; d < nvec; d += blockDim.x) {
        float4 v = trow[d];
        int base = d << 2;
        unsafeAtomicAdd(&arow[base + 0], w * v.x);
        unsafeAtomicAdd(&arow[base + 1], w * v.y);
        unsafeAtomicAdd(&arow[base + 2], w * v.z);
        unsafeAtomicAdd(&arow[base + 3], w * v.w);
    }
}

// ---------------------------------------------------------------------------
// host-side launch
// ---------------------------------------------------------------------------
extern "C" void kernel_launch(void* const* d_in, const int* in_sizes, int n_in,
                              void* d_out, int out_size, void* d_ws, size_t ws_size,
                              hipStream_t stream) {
    const float* x  = (const float*)d_in[0];
    const int*   ei = (const int*)d_in[1];
    const float* W1 = (const float*)d_in[2];
    const float* b1 = (const float*)d_in[3];
    const float* W2 = (const float*)d_in[4];
    const float* b2 = (const float*)d_in[5];

    const int D = in_sizes[3];                 // bias length
    const int n = in_sizes[0] / D;             // nodes
    const int E = in_sizes[1] / 2;             // edges
    const int* row = ei;
    const int* col = ei + E;

    // workspace layout (256B aligned)
    char* p = (char*)d_ws;
    size_t off = 0;
    auto take = [&](size_t bytes) -> char* {
        char* r = p + off;
        off = (off + bytes + 255) & ~(size_t)255;
        return r;
    };
    float*  dis = (float*)take((size_t)n * 4);
    float*  eb  = (float*)take((size_t)(D + 8) * 4);
    half_t* W16 = (half_t*)take((size_t)D * D * 2);
    float*  h32 = (float*)take((size_t)n * D * 4);   // doubles as agg
    half_t* h16 = (half_t*)take((size_t)n * D * 2);
    float*  mx  = (float*)take((size_t)n * D * 4);   // doubles as t

    // degree normalization
    deg_init_kernel<<<(n + 255) / 256, 256, 0, stream>>>(dis, n);
    deg_accum_kernel<<<(E + 255) / 256, 256, 0, stream>>>(col, dis, E);
    deg_to_dis_kernel<<<(n + 255) / 256, 256, 0, stream>>>(dis, n);

    // h = expmap0(x)
    expmap_rows_kernel<<<n, 256, 0, stream>>>(x, h32, h16, D);

    const int Ntiles2 = (D + 31) / 32;
    const int Mtiles2 = (n + 31) / 32;
    const int totalTiles2 = Mtiles2 * Ntiles2;
    const int gemmBlocks = (totalTiles2 + 7) / 8;    // 8 waves / block

    const float* Ws[2] = { W1, W2 };
    const float* bs[2] = { b1, b2 };

    for (int layer = 0; layer < 2; ++layer) {
        convert_f16_kernel<<<(D * D + 255) / 256, 256, 0, stream>>>(Ws[layer], W16, D * D);
        bias_expmap_kernel<<<1, 256, 0, stream>>>(bs[layer], eb, D);

        // mx = h @ W^T (f16 WMMA, f32 accumulate, 32x32 tile per wave)
        gemm_wmma_kernel<<<gemmBlocks, 256, 0, stream>>>(h16, W16, mx, n, D, D,
                                                         Ntiles2, totalTiles2);

        // t = logmap0(project(mobius_add(matvec_tail(mx), eb)))   (in place over mx)
        mobius_post_kernel<<<n, 256, 0, stream>>>(h32, mx, eb, D);

        // agg = segment_sum(norm * t[row], col)
        hipMemsetAsync(h32, 0, (size_t)n * D * 4, stream);
        aggregate_kernel<<<E + n, 128, 0, stream>>>(mx, row, col, dis, h32, E, D);

        // h = expmap0(agg)
        if (layer == 0)
            expmap_rows_kernel<<<n, 256, 0, stream>>>(h32, h32, h16, D);
        else
            expmap_rows_kernel<<<n, 256, 0, stream>>>(h32, (float*)d_out, (half_t*)0, D);
    }
}
